// MyAttn_1786706395393
// MI455X (gfx1250) — matrix operand order repaired and checked by reading, not compiled, not run on hardware
//
#include <hip/hip_runtime.h>
#include <hip/hip_bf16.h>

#define STATE_LEN 8192
#define SEQ_LEN   8192
#define HIDDEN    1024

typedef __bf16 bf16_t;
typedef bf16_t v16bf  __attribute__((ext_vector_type(16)));
typedef bf16_t bf16x8 __attribute__((ext_vector_type(8)));
typedef float  v8f    __attribute__((ext_vector_type(8)));
typedef float  f32x4  __attribute__((ext_vector_type(4)));

// ===========================================================================
// One-time f32 -> bf16 conversion of both matrices into workspace.
// Each thread converts 8 contiguous elements (2x b128 load, 4x cvt_pk, 1x b128 store).
// Grid covers A then B back-to-back.
// ===========================================================================
__global__ __launch_bounds__(256, 1)
void convert_to_bf16(const float* __restrict__ A, const float* __restrict__ B,
                     bf16_t* __restrict__ Abf, bf16_t* __restrict__ Bbf) {
  const size_t n   = (size_t)STATE_LEN * HIDDEN;
  const size_t idx = ((size_t)blockIdx.x * 256 + threadIdx.x) * 8;
  const float* src;
  bf16_t*      dst;
  size_t       off;
  if (idx < n) { src = A; dst = Abf; off = idx; }
  else         { src = B; dst = Bbf; off = idx - n; }
  f32x4 x0 = *(const f32x4*)(src + off);
  f32x4 x1 = *(const f32x4*)(src + off + 4);
  bf16x8 r;
#pragma unroll
  for (int i = 0; i < 4; ++i) {
    r[i]     = (bf16_t)x0[i];
    r[4 + i] = (bf16_t)x1[i];
  }
  *(bf16x8*)(dst + off) = r;
}

// ===========================================================================
// bf16 fragment loaders (fast path), per CDNA5 ISA 7.12.2 layouts.
// ===========================================================================
// A 16x32 (MxK): lane l row=l&15; kb=(l>>4)*8; halfs 0-7 = K[kb..+7],
// halfs 8-15 = K[kb+16..+23].  Two 16B loads, free concat.
__device__ __forceinline__ v16bf load_fragA_bf(const bf16_t* __restrict__ base,
                                               int row, int k0, int lane) {
  const bf16_t* p = base + (size_t)(row + (lane & 15)) * HIDDEN
                         + (k0 + ((lane >> 4) << 3));
  bf16x8 c0 = *(const bf16x8*)(p);
  bf16x8 c1 = *(const bf16x8*)(p + 16);
  return __builtin_shufflevector(c0, c1, 0, 1, 2, 3, 4, 5, 6, 7,
                                         8, 9, 10, 11, 12, 13, 14, 15);
}

// B 32x16 (KxN): lane l col=l&15; kb=(l>>4)*8; halfs 0-7 = K[kb..+7], halfs 8-15 = K[kb+16..+23].
// B[k][n] = history[n][k] -> a B column is a history row (contiguous).
__device__ __forceinline__ v16bf load_fragB_bf(const bf16_t* __restrict__ base,
                                               int row, int k0, int lane) {
  const bf16_t* p = base + (size_t)(row + (lane & 15)) * HIDDEN
                         + (k0 + ((lane >> 4) << 3));
  bf16x8 c0 = *(const bf16x8*)(p);
  bf16x8 c1 = *(const bf16x8*)(p + 16);
  return __builtin_shufflevector(c0, c1, 0, 1, 2, 3, 4, 5, 6, 7,
                                         8, 9, 10, 11, 12, 13, 14, 15);
}

// ===========================================================================
// f32 fragment loaders (fallback path, converts in-loop).
// ===========================================================================
__device__ __forceinline__ v16bf load_fragA_f32(const float* __restrict__ base,
                                                int row, int k0, int lane) {
  const float* p = base + (size_t)(row + (lane & 15)) * HIDDEN
                        + (k0 + ((lane >> 4) << 3));
  f32x4 a0 = *(const f32x4*)(p + 0);
  f32x4 a1 = *(const f32x4*)(p + 4);
  f32x4 b0 = *(const f32x4*)(p + 16);
  f32x4 b1 = *(const f32x4*)(p + 20);
  v16bf r;
#pragma unroll
  for (int i = 0; i < 4; ++i) {
    r[i]      = (bf16_t)a0[i];
    r[4 + i]  = (bf16_t)a1[i];
    r[8 + i]  = (bf16_t)b0[i];
    r[12 + i] = (bf16_t)b1[i];
  }
  return r;
}

__device__ __forceinline__ v16bf load_fragB_f32(const float* __restrict__ base,
                                                int row, int k0, int lane) {
  const float* p = base + (size_t)(row + (lane & 15)) * HIDDEN
                        + (k0 + ((lane >> 4) << 3));
  f32x4 c0 = *(const f32x4*)(p + 0);
  f32x4 c1 = *(const f32x4*)(p + 4);
  f32x4 c2 = *(const f32x4*)(p + 16);
  f32x4 c3 = *(const f32x4*)(p + 20);
  v16bf r;
#pragma unroll
  for (int i = 0; i < 4; ++i) {
    r[i]      = (bf16_t)c0[i];
    r[4 + i]  = (bf16_t)c1[i];
    r[8 + i]  = (bf16_t)c2[i];
    r[12 + i] = (bf16_t)c3[i];
  }
  return r;
}

// ===========================================================================
// Shared GEMM body: 128x128 block tile, 8 waves, wave tile 32x64 (2x4 WMMA).
// ===========================================================================
#define GEMM_BODY(LOAD_A, LOAD_B, APTR, BPTR)                                   \
  const int lane  = threadIdx.x & 31;                                           \
  const int wave  = threadIdx.x >> 5;                                           \
  const int tileM = blockIdx.y * 128 + (wave >> 1) * 32;                        \
  const int tileN = blockIdx.x * 128 + (wave & 1) * 64;                         \
  v8f acc[2][4];                                                                \
  _Pragma("unroll") for (int i = 0; i < 2; ++i)                                 \
    _Pragma("unroll") for (int j = 0; j < 4; ++j) acc[i][j] = (v8f)(0.0f);      \
  for (int k0 = 0; k0 < HIDDEN; k0 += 32) {                                     \
    v16bf a0 = LOAD_A(APTR, tileM,      k0, lane);                              \
    v16bf a1 = LOAD_A(APTR, tileM + 16, k0, lane);                              \
    v16bf b0 = LOAD_B(BPTR, tileN,      k0, lane);                              \
    v16bf b1 = LOAD_B(BPTR, tileN + 16, k0, lane);                              \
    v16bf b2 = LOAD_B(BPTR, tileN + 32, k0, lane);                              \
    v16bf b3 = LOAD_B(BPTR, tileN + 48, k0, lane);                              \
    acc[0][0] = __builtin_amdgcn_wmma_f32_16x16x32_bf16(false, a0, false, b0,   \
                    (short)0, acc[0][0], false, false);                         \
    acc[0][1] = __builtin_amdgcn_wmma_f32_16x16x32_bf16(false, a0, false, b1,   \
                    (short)0, acc[0][1], false, false);                         \
    acc[0][2] = __builtin_amdgcn_wmma_f32_16x16x32_bf16(false, a0, false, b2,   \
                    (short)0, acc[0][2], false, false);                         \
    acc[0][3] = __builtin_amdgcn_wmma_f32_16x16x32_bf16(false, a0, false, b3,   \
                    (short)0, acc[0][3], false, false);                         \
    acc[1][0] = __builtin_amdgcn_wmma_f32_16x16x32_bf16(false, a1, false, b0,   \
                    (short)0, acc[1][0], false, false);                         \
    acc[1][1] = __builtin_amdgcn_wmma_f32_16x16x32_bf16(false, a1, false, b1,   \
                    (short)0, acc[1][1], false, false);                         \
    acc[1][2] = __builtin_amdgcn_wmma_f32_16x16x32_bf16(false, a1, false, b2,   \
                    (short)0, acc[1][2], false, false);                         \
    acc[1][3] = __builtin_amdgcn_wmma_f32_16x16x32_bf16(false, a1, false, b3,   \
                    (short)0, acc[1][3], false, false);                         \
  }                                                                             \
  const int col  = lane & 15;                                                   \
  const int rofs = (lane >> 4) * 8;                                             \
  _Pragma("unroll") for (int mi = 0; mi < 2; ++mi) {                            \
    _Pragma("unroll") for (int ni = 0; ni < 4; ++ni) {                          \
      const int rowBase = tileM + mi * 16 + rofs;                               \
      const int c = tileN + ni * 16 + col;                                      \
      _Pragma("unroll") for (int r = 0; r < 8; ++r) {                           \
        out[(size_t)(rowBase + r) * SEQ_LEN + c] = acc[mi][ni][r];              \
      }                                                                         \
    }                                                                           \
  }

// Fast path: pre-converted bf16 inputs from workspace.
__global__ __launch_bounds__(256, 1)
void attn_energies_wmma_bf16(const bf16_t* __restrict__ A,
                             const bf16_t* __restrict__ B,
                             float* __restrict__ out) {
  GEMM_BODY(load_fragA_bf, load_fragB_bf, A, B)
}

// Fallback path: f32 inputs, convert in the loop (if ws too small).
__global__ __launch_bounds__(256, 1)
void attn_energies_wmma_f32(const float* __restrict__ A,
                            const float* __restrict__ B,
                            float* __restrict__ out) {
  GEMM_BODY(load_fragA_f32, load_fragB_f32, A, B)
}

// ===========================================================================
// Row softmax, in place. One 256-thread block per row; 32 elems/thread.
// ===========================================================================
__global__ __launch_bounds__(256, 1)
void row_softmax(float* __restrict__ out) {
  const int row  = blockIdx.x;
  const int tid  = threadIdx.x;
  const int lane = tid & 31;
  const int wave = tid >> 5;
  float* p = out + (size_t)row * SEQ_LEN;

  float v[32];
  float m = -__builtin_inff();
#pragma unroll
  for (int i = 0; i < 32; ++i) {
    v[i] = p[tid + i * 256];
    m = fmaxf(m, v[i]);
  }
#pragma unroll
  for (int off = 16; off > 0; off >>= 1)
    m = fmaxf(m, __shfl_xor(m, off, 32));

  __shared__ float smax[8];
  __shared__ float ssum[8];
  if (lane == 0) smax[wave] = m;
  __syncthreads();
  float rmax = smax[0];
#pragma unroll
  for (int i = 1; i < 8; ++i) rmax = fmaxf(rmax, smax[i]);

  float s = 0.0f;
#pragma unroll
  for (int i = 0; i < 32; ++i) {
    v[i] = __expf(v[i] - rmax);   // v_exp_f32
    s += v[i];
  }
#pragma unroll
  for (int off = 16; off > 0; off >>= 1)
    s += __shfl_xor(s, off, 32);
  if (lane == 0) ssum[wave] = s;
  __syncthreads();
  float rsum = 0.0f;
#pragma unroll
  for (int i = 0; i < 8; ++i) rsum += ssum[i];

  const float inv = 1.0f / rsum;
#pragma unroll
  for (int i = 0; i < 32; ++i)
    p[tid + i * 256] = v[i] * inv;
}

// ===========================================================================
extern "C" void kernel_launch(void* const* d_in, const int* in_sizes, int n_in,
                              void* d_out, int out_size, void* d_ws, size_t ws_size,
                              hipStream_t stream) {
  (void)in_sizes; (void)n_in; (void)out_size;
  const float* out_state = (const float*)d_in[0];  // [STATE_LEN, HIDDEN]
  const float* history   = (const float*)d_in[1];  // [SEQ_LEN,  HIDDEN]
  float* out = (float*)d_out;                      // [STATE_LEN, SEQ_LEN]

  const size_t elems    = (size_t)STATE_LEN * HIDDEN;
  const size_t ws_need  = 2 * elems * sizeof(bf16_t);   // 32 MB
  dim3 grid(SEQ_LEN / 128, STATE_LEN / 128);

  if (ws_size >= ws_need) {
    bf16_t* Abf = (bf16_t*)d_ws;
    bf16_t* Bbf = Abf + elems;
    const int cvt_blocks = (int)((2 * elems / 8) / 256);  // 8192
    convert_to_bf16<<<cvt_blocks, 256, 0, stream>>>(out_state, history, Abf, Bbf);
    attn_energies_wmma_bf16<<<grid, 256, 0, stream>>>(Abf, Bbf, out);
  } else {
    attn_energies_wmma_f32<<<grid, 256, 0, stream>>>(out_state, history, out);
  }
  row_softmax<<<STATE_LEN, 256, 0, stream>>>(out);
}
